// DeltaCorrection_85899345920397
// MI455X (gfx1250) — matrix operand order repaired and checked by reading, not compiled, not run on hardware
//
#include <hip/hip_runtime.h>

// Delta-correction chunked linear attention for MI455X (gfx1250, wave32).
// B=4, H=16, N=4096, D=64, C=64.  One workgroup per (b,h); 16 waves; each
// wave owns one 16x16 tile of every 64x64 GEMM.  bf16 WMMA, f32 accumulate.
// All WMMA fragment gathers are contiguous (2x ds_load_b128 each) via
// transposed LDS mirrors (vbT, kb2T) padded to ld=72.

typedef __bf16 bf16_t;
typedef bf16_t v16bf __attribute__((ext_vector_type(16)));
typedef bf16_t v4bf  __attribute__((ext_vector_type(4)));
typedef float  v8f   __attribute__((ext_vector_type(8)));

#define THREADS 512
#define CHUNK   64
#define DIM     64
#define NCHUNKS 64
#define CELEMS  (CHUNK * DIM)   // 4096 elements per chunk per tensor
#define LDT     72              // padded row stride for transposed mirrors

// ---- WMMA fragment lane maps (CDNA5 ISA 7.12.2, 16-bit A 16x32 / B 32x16) --
__device__ __forceinline__ int amap_k(int lane, int e) {
  // lanes 0-15: K = 0..7 (e<8), 16..23 (e>=8); lanes 16-31: K = 8..15, 24..31
  return ((e < 8) ? ((lane >= 16) ? 8 : 0) : ((lane >= 16) ? 24 : 16)) + (e & 7);
}
__device__ __forceinline__ int bmap_k(int lane, int e) {
  // lanes 0-15: K = 0..7, 16..23 ; lanes 16-31: K = 8..15, 24..31
  return ((e < 8) ? ((lane >= 16) ? 8 : 0) : ((lane >= 16) ? 24 : 16)) + (e & 7);
}

// A[m][k] from row-major src (two contiguous 8-element segments per lane)
__device__ __forceinline__ v16bf load_A_rm(const bf16_t* __restrict__ src,
                                           int ld, int row0, int k0, int lane) {
  v16bf a;
  const int r = row0 + (lane & 15);
#pragma unroll
  for (int e = 0; e < 16; ++e) a[e] = src[r * ld + k0 + amap_k(lane, e)];
  return a;
}
// B[k][n] = src[n*ld + k]  (two contiguous 8-element segments per lane)
__device__ __forceinline__ v16bf load_B_tr(const bf16_t* __restrict__ src,
                                           int ld, int k0, int col0, int lane) {
  v16bf b;
  const int c = col0 + (lane & 15);
#pragma unroll
  for (int e = 0; e < 16; ++e) b[e] = src[c * ld + k0 + bmap_k(lane, e)];
  return b;
}

__device__ __forceinline__ v8f wmma_bf16(v16bf a, v16bf b, v8f c) {
  return __builtin_amdgcn_wmma_f32_16x16x32_bf16(
      /*neg_a=*/false, a, /*neg_b=*/false, b,
      /*c_mod=*/(short)0, c, /*reuse_a=*/false, /*reuse_b=*/false);
}

__global__ void __launch_bounds__(THREADS)
delta_correction_kernel(const float* __restrict__ q,
                        const float* __restrict__ k,
                        const float* __restrict__ v,
                        const float* __restrict__ Wd, const float* __restrict__ bd,
                        const float* __restrict__ Ww, const float* __restrict__ bw,
                        const float* __restrict__ osc,
                        float* __restrict__ out) {
  // kf32 (f32 staging for k) is reused as the bf16 scores buffer; the two
  // uses live in disjoint phases separated by barriers.
  __shared__ __align__(16) float  kf32[CELEMS];        // 16 KB (alias: scores)
  __shared__ __align__(16) bf16_t qb[CELEMS];          // 8 KB   q (row major)
  __shared__ __align__(16) bf16_t kb2[CELEMS];         // 8 KB   beta*k/||k||
  __shared__ __align__(16) bf16_t vbT[DIM * LDT];      // 9 KB   v transposed
  __shared__ __align__(16) bf16_t kb2T[DIM * LDT];     // 9 KB   kb2 transposed
  __shared__ __align__(16) bf16_t Sbf[DIM * DIM];      // 8 KB   state mirror
  __shared__ float s_Ww[DIM], s_Wd[DIM], s_alpha[CHUNK];
  __shared__ float s_decay;
  bf16_t* sc = (bf16_t*)kf32;                          // scores alias

  const int bh   = blockIdx.x;                         // 0..63 == (b*H + h)
  const int tid  = threadIdx.x;
  const int lane = tid & 31;
  const int wave = tid >> 5;                           // 0..15
  const int tm   = (wave >> 2) * 16;                   // tile row origin
  const int tn   = (wave & 3) * 16;                    // tile col origin
  const int col  = tn + (lane & 15);
  const int rb   = tm + ((lane >= 16) ? 8 : 0);

  const float bwv = bw[0], bdv = bd[0], oscale = osc[0];

  if (tid < DIM) { s_Ww[tid] = Ww[tid]; s_Wd[tid] = Wd[tid]; }
  for (int i = tid; i < DIM * DIM; i += THREADS) Sbf[i] = (bf16_t)0.0f;
  v8f Sreg = {0.f, 0.f, 0.f, 0.f, 0.f, 0.f, 0.f, 0.f}; // this wave's S tile
  __syncthreads();

  const size_t base = (size_t)bh * (size_t)(NCHUNKS * CELEMS);

  for (int ch = 0; ch < NCHUNKS; ++ch) {
    const float* qg = q + base + (size_t)ch * CELEMS;
    const float* kg = k + base + (size_t)ch * CELEMS;
    const float* vg = v + base + (size_t)ch * CELEMS;

    // Prefetch next chunk while this one is computed (global_prefetch_b8).
    if (ch + 1 < NCHUNKS) {
      const int po = CELEMS + tid * 8;
      __builtin_prefetch((const void*)(qg + po), 0, 1);
      __builtin_prefetch((const void*)(kg + po), 0, 1);
      __builtin_prefetch((const void*)(vg + po), 0, 1);
    }

    // ---- Stage chunk into LDS: q->bf16, k->f32, v->bf16 transposed --------
    {
      const float4* q4 = (const float4*)qg;
      const float4* k4 = (const float4*)kg;
      const float4* v4 = (const float4*)vg;
#pragma unroll
      for (int it = 0; it < CELEMS / 4 / THREADS; ++it) {
        const int i = tid + it * THREADS;
        const int o = i * 4;
        float4 tq = q4[i];
        v4bf pq = {(bf16_t)tq.x, (bf16_t)tq.y, (bf16_t)tq.z, (bf16_t)tq.w};
        *(v4bf*)&qb[o] = pq;                            // one b64 store
        *(float4*)&kf32[o] = k4[i];                     // one b128 store
        float4 tv = v4[i];
        const int r = o >> 6, c = o & 63;               // v[r][c..c+3]
        vbT[(c + 0) * LDT + r] = (bf16_t)tv.x;
        vbT[(c + 1) * LDT + r] = (bf16_t)tv.y;
        vbT[(c + 2) * LDT + r] = (bf16_t)tv.z;
        vbT[(c + 3) * LDT + r] = (bf16_t)tv.w;
      }
    }
    __syncthreads();

    // ---- Gates + key normalization: k2 = sigmoid(k.Ww+bw) * k/||k|| -------
    if (tid < CHUNK) {
      const float4* kr4 = (const float4*)&kf32[tid * DIM];
      float nrm = 0.f, dw = 0.f, dd = 0.f;
#pragma unroll
      for (int j4 = 0; j4 < DIM / 4; ++j4) {
        const float4 x  = kr4[j4];
        const float4 w  = *(const float4*)&s_Ww[j4 * 4];
        const float4 wd = *(const float4*)&s_Wd[j4 * 4];
        nrm += x.x * x.x + x.y * x.y + x.z * x.z + x.w * x.w;
        dw  += x.x * w.x + x.y * w.y + x.z * w.z + x.w * w.w;
        dd  += x.x * wd.x + x.y * wd.y + x.z * wd.z + x.w * wd.w;
      }
      const float inv   = 1.0f / fmaxf(sqrtf(nrm), 1e-12f);
      const float beta  = 1.0f / (1.0f + __expf(-(dw + bwv)));
      const float alpha = 1.0f / (1.0f + __expf(-(dd + bdv)));
      s_alpha[tid] = alpha;
      const float s = beta * inv;
#pragma unroll
      for (int j4 = 0; j4 < DIM / 4; ++j4) {
        const float4 x = kr4[j4];
        v4bf p = {(bf16_t)(x.x * s), (bf16_t)(x.y * s),
                  (bf16_t)(x.z * s), (bf16_t)(x.w * s)};
        *(v4bf*)&kb2[tid * DIM + j4 * 4] = p;
      }
    }
    __syncthreads();

    // ---- kb2T transpose (all threads) + chunk decay -----------------------
    {
      const int r  = tid >> 3;                 // 8 threads per row
      const int c0 = (tid & 7) * 8;            // 8 columns each
      v4bf a = *(const v4bf*)&kb2[r * DIM + c0];
      v4bf b = *(const v4bf*)&kb2[r * DIM + c0 + 4];
#pragma unroll
      for (int j = 0; j < 4; ++j) {
        kb2T[(c0 + j) * LDT + r]     = a[j];
        kb2T[(c0 + 4 + j) * LDT + r] = b[j];
      }
    }
    if (tid == 0) {  // chunk decay = mean(alpha)^C
      float m = 0.f;
#pragma unroll 8
      for (int i = 0; i < CHUNK; ++i) m += s_alpha[i];
      s_decay = __powf(m * (1.0f / CHUNK), (float)CHUNK);
    }

    // ---- GEMM1: scores = q @ k2^T, causal mask, -> bf16 scores ------------
    {
      v8f acc = {0.f, 0.f, 0.f, 0.f, 0.f, 0.f, 0.f, 0.f};
#pragma unroll
      for (int k0 = 0; k0 < DIM; k0 += 32) {
        v16bf a = load_A_rm(qb, DIM, tm, k0, lane);
        v16bf b = load_B_tr(kb2, DIM, k0, tn, lane);
        acc = wmma_bf16(a, b, acc);
      }
#pragma unroll
      for (int g = 0; g < 8; ++g) {
        const int row = rb + g;
        const float x = (col <= row) ? acc[g] : 0.0f;
        sc[row * DIM + col] = (bf16_t)x;       // overwrites kf32 (done w/ it)
      }
    }
    __syncthreads();

    // ---- GEMM2 (intra = scores @ v) + GEMM3 (inter = q @ S^T) -------------
    {
      v8f oacc = {0.f, 0.f, 0.f, 0.f, 0.f, 0.f, 0.f, 0.f};
#pragma unroll
      for (int k0 = 0; k0 < CHUNK; k0 += 32) {
        v16bf a = load_A_rm(sc, DIM, tm, k0, lane);
        v16bf b = load_B_tr(vbT, LDT, k0, tn, lane);   // B[c][n] = v[c][n]
        oacc = wmma_bf16(a, b, oacc);
      }
#pragma unroll
      for (int k0 = 0; k0 < DIM; k0 += 32) {
        v16bf a = load_A_rm(qb, DIM, tm, k0, lane);
        v16bf b = load_B_tr(Sbf, DIM, k0, tn, lane);   // B[j][i] = S[i][j]
        oacc = wmma_bf16(a, b, oacc);
      }
      float* og = out + base + (size_t)ch * CELEMS;
#pragma unroll
      for (int g = 0; g < 8; ++g)
        og[(rb + g) * DIM + col] = oacc[g] * oscale;
    }
    __syncthreads();  // all waves done reading Sbf before it is updated

    // ---- GEMM4: kv = v^T @ k2 ; S = decay*S + kv --------------------------
    {
      v8f kvacc = {0.f, 0.f, 0.f, 0.f, 0.f, 0.f, 0.f, 0.f};
#pragma unroll
      for (int k0 = 0; k0 < CHUNK; k0 += 32) {
        v16bf a = load_A_rm(vbT, LDT, tm, k0, lane);   // A[i][c] = v[c][i]
        v16bf b = load_B_tr(kb2T, LDT, k0, tn, lane);  // B[c][j] = kb2[c][j]
        kvacc = wmma_bf16(a, b, kvacc);
      }
      const float dcy = s_decay;
#pragma unroll
      for (int g = 0; g < 8; ++g) {
        const float sn = dcy * Sreg[g] + kvacc[g];
        Sreg[g] = sn;
        Sbf[(rb + g) * DIM + col] = (bf16_t)sn;
      }
    }
    __syncthreads();  // S mirror ready; LDS staging may be overwritten
  }
}

extern "C" void kernel_launch(void* const* d_in, const int* in_sizes, int n_in,
                              void* d_out, int out_size, void* d_ws, size_t ws_size,
                              hipStream_t stream) {
  const float* q   = (const float*)d_in[0];
  const float* k   = (const float*)d_in[1];
  const float* v   = (const float*)d_in[2];
  const float* Wd  = (const float*)d_in[3];
  const float* bd  = (const float*)d_in[4];
  const float* Ww  = (const float*)d_in[5];
  const float* bw  = (const float*)d_in[6];
  const float* osc = (const float*)d_in[7];
  float* out = (float*)d_out;

  (void)in_sizes; (void)n_in; (void)out_size; (void)d_ws; (void)ws_size;

  delta_correction_kernel<<<dim3(64), dim3(THREADS), 0, stream>>>(
      q, k, v, Wd, bd, Ww, bw, osc, out);
}
